// MixtureOfExperts_35888746725659
// MI455X (gfx1250) — matrix operand order repaired and checked by reading, not compiled, not run on hardware
//
#include <hip/hip_runtime.h>
#include <hip/hip_bf16.h>
#include <math.h>
#include <stdint.h>

// Problem constants (from reference): B=2, S=1024 -> T=2048 tokens
#define T_TOK 2048
#define DDIM  1024
#define FDIM  4096
#define NEXP  8

typedef __attribute__((ext_vector_type(16))) __bf16       v16bf;
typedef __attribute__((ext_vector_type(8)))  float        v8f;
typedef __attribute__((ext_vector_type(4)))  unsigned int v4u;
typedef __attribute__((ext_vector_type(4)))  unsigned int u32x4;
typedef __attribute__((ext_vector_type(8)))  int          i32x8;
typedef __attribute__((ext_vector_type(4)))  int          i32x4;

union Frag16 { v4u u[2]; v16bf v; };

#ifndef MOE_HAVE_TDM
#if defined(__has_builtin)
#if __has_builtin(__builtin_amdgcn_tensor_load_to_lds)
#define MOE_HAVE_TDM 1
#else
#define MOE_HAVE_TDM 0
#endif
#else
#define MOE_HAVE_TDM 0
#endif
#endif

static __device__ inline float gelu_erf(float x) {
  return 0.5f * x * (1.0f + erff(x * 0.7071067811865476f));
}

#if MOE_HAVE_TDM
// ---------------------------------------------------------------------------
// TDM 2-D tile load: 128 rows x 32 bf16 from an N-major weight matrix into
// LDS with hardware padding (16 dwords data + 4 dwords pad = 80B row stride,
// matching the bank-conflict-free fragment layout).  Descriptor packing per
// CDNA5 ISA.  Issued by one wave; tracked by TENSORcnt.
// ---------------------------------------------------------------------------
static __device__ inline void tdm_load_b_tile(const __hip_bfloat16* gsrc,
                                              unsigned lds_byte, int K, int N) {
  unsigned long long ga = (unsigned long long)(uintptr_t)gsrc;
  u32x4 g0;
  g0[0] = 1u;                                              // count=1 (valid), user mode
  g0[1] = lds_byte;                                        // lds_addr [63:32]
  g0[2] = (unsigned)(ga & 0xFFFFFFFFu);                    // global_addr lo
  g0[3] = (unsigned)((ga >> 32) & 0x01FFFFFFu) | (2u << 30); // addr hi | type=2
  i32x8 g1;
  g1[0] = (int)((1u << 16)      // data_size = 2 bytes
              | (1u << 20)      // pad_enable
              | (3u << 22)      // pad_interval: 16 dwords (row = 32 bf16 = 64B)
              | (3u << 25));    // pad_amount: 4 dwords (16B) -> 80B LDS stride
  g1[1] = (int)((unsigned)K << 16);                              // tensor_dim0[15:0]
  g1[2] = (int)((((unsigned)K >> 16) & 0xFFFFu) | ((unsigned)N << 16)); // dim0 hi | dim1 lo
  g1[3] = (int)((((unsigned)N >> 16) & 0xFFFFu) | (32u << 16));  // dim1 hi | tile_dim0=32
  g1[4] = (int)128u;                                             // tile_dim1=128, tile_dim2=0
  g1[5] = (int)(unsigned)K;                                      // tensor_dim0_stride lo32
  g1[6] = 0;                                                     // stride0 hi | stride1 lo
  g1[7] = 0;
  i32x4 z4 = {0, 0, 0, 0};
#if defined(__clang_major__) && (__clang_major__ >= 23)
  i32x8 z8 = {0, 0, 0, 0, 0, 0, 0, 0};
  __builtin_amdgcn_tensor_load_to_lds(g0, g1, z4, z4, z8, 0);
#else
  __builtin_amdgcn_tensor_load_to_lds(g0, g1, z4, z4, 0);
#endif
}
#endif  // MOE_HAVE_TDM

// ---------------------------------------------------------------------------
// 1) Router: logits = x @ W_router, softmax -> top2 -> renormalize.
//    top2 renorm of softmax == softmax over the two top logits.
// ---------------------------------------------------------------------------
__global__ __launch_bounds__(256) void moe_router(
    const float* __restrict__ x, const float* __restrict__ Wr,
    int* __restrict__ tok_e, float* __restrict__ tok_w) {
  int t = blockIdx.x;
  int tid = threadIdx.x;
  int e = tid & 7, seg = tid >> 3;              // 8 experts x 32 segments
  const float* xr = x + (size_t)t * DDIM;
  float p = 0.f;
  for (int k = seg; k < DDIM; k += 32) p += xr[k] * Wr[k * NEXP + e];
  __shared__ float red[256];
  red[tid] = p;
  __syncthreads();
  for (int s = 16; s > 0; s >>= 1) {
    if (seg < s) red[tid] += red[tid + 8 * s];
    __syncthreads();
  }
  if (tid == 0) {
    float l[8];
    #pragma unroll
    for (int i = 0; i < 8; ++i) l[i] = red[i];
    int i1 = 0;
    for (int i = 1; i < 8; ++i) if (l[i] > l[i1]) i1 = i;
    int i2 = (i1 == 0) ? 1 : 0;
    for (int i = 0; i < 8; ++i) if (i != i1 && l[i] > l[i2]) i2 = i;
    float e2 = expf(l[i2] - l[i1]);
    float w1 = 1.f / (1.f + e2);
    tok_e[2 * t]     = i1; tok_e[2 * t + 1] = i2;
    tok_w[2 * t]     = w1; tok_w[2 * t + 1] = 1.f - w1;
  }
}

// ---------------------------------------------------------------------------
// 2) Deterministic per-expert slot assignment (no atomics -> replay stable).
// ---------------------------------------------------------------------------
__global__ void moe_build(const int* __restrict__ tok_e,
                          int* __restrict__ off, int* __restrict__ cnt,
                          int* __restrict__ slot_token, int* __restrict__ tok2slot) {
  int e = threadIdx.x;  // 0..7
  int c = 0;
  for (int t = 0; t < T_TOK; ++t) {
    if (tok_e[2 * t]     == e) c++;
    if (tok_e[2 * t + 1] == e) c++;
  }
  cnt[e] = c;
  __syncthreads();
  if (e == 0) {
    int o = 0;
    for (int i = 0; i < NEXP; ++i) { off[i] = o; o += cnt[i]; }
  }
  __syncthreads();
  int o = off[e], r = 0;
  for (int t = 0; t < T_TOK; ++t)
    for (int j = 0; j < 2; ++j)
      if (tok_e[2 * t + j] == e) { slot_token[o + r] = t; tok2slot[2 * t + j] = o + r; r++; }
}

// ---------------------------------------------------------------------------
// 3) Weight transpose + fp32->bf16 convert.  in:[R][C] -> out:[C][R] per expert.
// ---------------------------------------------------------------------------
__global__ __launch_bounds__(256) void transpose_cvt_bf16(
    const float* __restrict__ in, __hip_bfloat16* __restrict__ out, int R, int C) {
  __shared__ float tile[32][33];
  size_t ebase = (size_t)blockIdx.z * (size_t)R * (size_t)C;
  const float* inE = in + ebase;
  __hip_bfloat16* outE = out + ebase;
  int c0 = blockIdx.x * 32, r0 = blockIdx.y * 32;
  int tx = threadIdx.x, ty = threadIdx.y;  // 32 x 8
  #pragma unroll
  for (int i = 0; i < 32; i += 8)
    tile[ty + i][tx] = inE[(size_t)(r0 + ty + i) * C + (c0 + tx)];
  __syncthreads();
  #pragma unroll
  for (int i = 0; i < 32; i += 8)
    outE[(size_t)(c0 + ty + i) * R + (r0 + tx)] = __float2bfloat16(tile[tx][ty + i]);
}

// ---------------------------------------------------------------------------
// 4) Gather routed tokens into compact bf16 activation matrix Xc[2T][D].
// ---------------------------------------------------------------------------
__global__ __launch_bounds__(256) void moe_gather(
    const float* __restrict__ x, const int* __restrict__ slot_token,
    __hip_bfloat16* __restrict__ Xc) {
  int s = blockIdx.x;
  int t = slot_token[s];
  const float* xr = x + (size_t)t * DDIM;
  __hip_bfloat16* xo = Xc + (size_t)s * DDIM;
  for (int i = threadIdx.x; i < DDIM; i += 256) xo[i] = __float2bfloat16(xr[i]);
}

// ---------------------------------------------------------------------------
// 5) WMMA bf16 GEMM over a compact expert region, software-pipelined:
//    double-buffered LDS; B weight tiles DMA'd by the Tensor Data Mover
//    (tensor_load_to_lds + s_wait_tensorcnt), A tiles via predicated b128
//    loads (zero-fill past rowsValid).  Block tile 128x128, K-step 32,
//    8 wave32 waves each computing a 64x32 sub-tile (4x2 WMMA frags).
//    mode 0: +bias, gelu -> bf16 store   (GEMM1)
//    mode 1: +bias -> f32 store          (GEMM2)
// ---------------------------------------------------------------------------
#define LDS_STRIDE 40          // halves: 64B data + 16B pad = 80B rows
#define SZH (128 * LDS_STRIDE) // halves per tile buffer

__global__ __launch_bounds__(256) void moe_gemm_wmma(
    const __hip_bfloat16* __restrict__ A,    // [2T][K] compact rows
    const __hip_bfloat16* __restrict__ Bt,   // [NEXP][N][K] (N-major)
    const float* __restrict__ bias,          // [NEXP][N]
    void* __restrict__ Cout,                 // [2T][N]
    const int* __restrict__ off, const int* __restrict__ cnt,
    int K, int N, int mode) {
  int e = blockIdx.z;
  int cntE = cnt[e];
  int m0 = blockIdx.y * 128;
  if (m0 >= cntE) return;                    // data-dependent early exit
  int rowsValid = cntE - m0; if (rowsValid > 128) rowsValid = 128;
  int offE = off[e];
  int nBase = blockIdx.x * 128;

  const __hip_bfloat16* Aexp = A + (size_t)(offE + m0) * (size_t)K;
  const __hip_bfloat16* Bexp = Bt + (size_t)e * (size_t)N * (size_t)K;
  const __hip_bfloat16* Bcol = Bexp + (size_t)nBase * (size_t)K;
  const float* biasE = bias + (size_t)e * (size_t)N;

  __shared__ __align__(16) __hip_bfloat16 sA[2 * SZH];
  __shared__ __align__(16) __hip_bfloat16 sB[2 * SZH];

  int tid  = threadIdx.x;
  int lane = tid & 31;
  int wave = tid >> 5;
  int wm = wave & 1;   // M sub-tile (0..1) -> 64 rows
  int wn = wave >> 1;  // N sub-tile (0..3) -> 32 cols

  // Each thread owns two 16B chunks of the 128x32 tile fill.
  int rowA0 = tid >> 2, rowA1 = rowA0 + 64;
  int qA = (tid & 3) * 8;
  const v4u zero4 = {0u, 0u, 0u, 0u};

#if MOE_HAVE_TDM
  unsigned ldsB0 = (unsigned)(uintptr_t)(void*)&sB[0];
  unsigned ldsB1 = (unsigned)(uintptr_t)(void*)&sB[SZH];
#endif

  const v8f vzero = {0.f, 0.f, 0.f, 0.f, 0.f, 0.f, 0.f, 0.f};
  v8f acc[4][2];
  #pragma unroll
  for (int mi = 0; mi < 4; ++mi)
    #pragma unroll
    for (int ni = 0; ni < 2; ++ni) acc[mi][ni] = vzero;

  // ---- prologue: stage K-tile 0 into buffer 0 ----
  {
    v4u r0 = zero4, r1 = zero4;
    if (rowA0 < rowsValid) r0 = *(const v4u*)(Aexp + (size_t)rowA0 * K + qA);
    if (rowA1 < rowsValid) r1 = *(const v4u*)(Aexp + (size_t)rowA1 * K + qA);
#if MOE_HAVE_TDM
    if (wave == 0) tdm_load_b_tile(Bcol, ldsB0, K, N);
#else
    {
      v4u w0 = *(const v4u*)(Bcol + (size_t)rowA0 * K + qA);
      v4u w1 = *(const v4u*)(Bcol + (size_t)rowA1 * K + qA);
      *(v4u*)(sB + rowA0 * LDS_STRIDE + qA) = w0;
      *(v4u*)(sB + rowA1 * LDS_STRIDE + qA) = w1;
    }
#endif
    *(v4u*)(sA + rowA0 * LDS_STRIDE + qA) = r0;
    *(v4u*)(sA + rowA1 * LDS_STRIDE + qA) = r1;
#if MOE_HAVE_TDM
    if (wave == 0) __builtin_amdgcn_s_wait_tensorcnt((short)0);
#endif
    __syncthreads();
  }

  int nk = K >> 5;
  for (int it = 0; it < nk; ++it) {
    int cur = it & 1;
    const __hip_bfloat16* curA = sA + cur * SZH;
    const __hip_bfloat16* curB = sB + cur * SZH;

    // ---- stage K-tile it+1 into the other buffer (overlaps WMMA below) ----
    if (it + 1 < nk) {
      int ktn = (it + 1) << 5;
      v4u r0 = zero4, r1 = zero4;
      if (rowA0 < rowsValid) r0 = *(const v4u*)(Aexp + (size_t)rowA0 * K + ktn + qA);
      if (rowA1 < rowsValid) r1 = *(const v4u*)(Aexp + (size_t)rowA1 * K + ktn + qA);
#if MOE_HAVE_TDM
      if (wave == 0) tdm_load_b_tile(Bcol + ktn, cur ? ldsB0 : ldsB1, K, N);
#else
      {
        __hip_bfloat16* nB = sB + (cur ^ 1) * SZH;
        v4u w0 = *(const v4u*)(Bcol + (size_t)rowA0 * K + ktn + qA);
        v4u w1 = *(const v4u*)(Bcol + (size_t)rowA1 * K + ktn + qA);
        *(v4u*)(nB + rowA0 * LDS_STRIDE + qA) = w0;
        *(v4u*)(nB + rowA1 * LDS_STRIDE + qA) = w1;
      }
#endif
      __hip_bfloat16* nA = sA + (cur ^ 1) * SZH;
      *(v4u*)(nA + rowA0 * LDS_STRIDE + qA) = r0;
      *(v4u*)(nA + rowA1 * LDS_STRIDE + qA) = r1;
    }

    // ---- compute on current buffers ----
    // B fragment: lane = column n; VGPR j holds K = kh*16 + 2j,2j+1.
    Frag16 b0, b1f, af;
    {
      int n = lane & 15, kh = lane >> 4;
      const __hip_bfloat16* pb0 = curB + (wn * 32 + n) * LDS_STRIDE + kh * 16;
      b0.u[0]  = *(const v4u*)(pb0);
      b0.u[1]  = *(const v4u*)(pb0 + 8);
      const __hip_bfloat16* pb1 = curB + (wn * 32 + 16 + n) * LDS_STRIDE + kh * 16;
      b1f.u[0] = *(const v4u*)(pb1);
      b1f.u[1] = *(const v4u*)(pb1 + 8);
    }
    // A fragment: lane = row m; VGPR j holds K = kh*16 + 2j,2j+1 (matches B).
    #pragma unroll
    for (int mi = 0; mi < 4; ++mi) {
      int m = lane & 15, kh = lane >> 4;
      const __hip_bfloat16* pa = curA + (wm * 64 + mi * 16 + m) * LDS_STRIDE + kh * 16;
      af.u[0] = *(const v4u*)(pa);
      af.u[1] = *(const v4u*)(pa + 8);
      acc[mi][0] = __builtin_amdgcn_wmma_f32_16x16x32_bf16(
          false, af.v, false, b0.v,  (short)0, acc[mi][0], false, false);
      acc[mi][1] = __builtin_amdgcn_wmma_f32_16x16x32_bf16(
          false, af.v, false, b1f.v, (short)0, acc[mi][1], false, false);
    }

#if MOE_HAVE_TDM
    if (wave == 0) __builtin_amdgcn_s_wait_tensorcnt((short)0);
#endif
    __syncthreads();  // publishes next buffers; also fences ds stores
  }

  // ---- epilogue. C/D layout: lane n = lane&15; VGPR r -> M = r + 8*(lane>>4).
  int nl = lane & 15, mh = lane >> 4;
  __hip_bfloat16* Cb = (__hip_bfloat16*)Cout;
  float* Cf = (float*)Cout;
  #pragma unroll
  for (int mi = 0; mi < 4; ++mi) {
    #pragma unroll
    for (int ni = 0; ni < 2; ++ni) {
      int col = nBase + wn * 32 + ni * 16 + nl;
      float bv = biasE[col];
      #pragma unroll
      for (int r = 0; r < 8; ++r) {
        int rowT = wm * 64 + mi * 16 + mh * 8 + r;
        if (rowT < rowsValid) {
          float v = acc[mi][ni][r] + bv;
          size_t gi = (size_t)(offE + m0 + rowT) * (size_t)N + col;
          if (mode == 0) Cb[gi] = __float2bfloat16(gelu_erf(v));
          else           Cf[gi] = v;
        }
      }
    }
  }
}

// ---------------------------------------------------------------------------
// 6) Weighted combine: out[t] = w0*Y[slot0] + w1*Y[slot1]  (no atomics).
// ---------------------------------------------------------------------------
__global__ __launch_bounds__(256) void moe_combine(
    const float* __restrict__ Yc, const int* __restrict__ tok2slot,
    const float* __restrict__ tok_w, float* __restrict__ out) {
  int t = blockIdx.x;
  int s0 = tok2slot[2 * t], s1 = tok2slot[2 * t + 1];
  float w0 = tok_w[2 * t], w1 = tok_w[2 * t + 1];
  const float* y0 = Yc + (size_t)s0 * DDIM;
  const float* y1 = Yc + (size_t)s1 * DDIM;
  float* o = out + (size_t)t * DDIM;
  for (int i = threadIdx.x; i < DDIM; i += 256)
    o[i] = w0 * y0[i] + w1 * y1[i];
}

// ---------------------------------------------------------------------------
extern "C" void kernel_launch(void* const* d_in, const int* in_sizes, int n_in,
                              void* d_out, int out_size, void* d_ws, size_t ws_size,
                              hipStream_t stream) {
  const float* x  = (const float*)d_in[0];  // [T, D]
  const float* Wr = (const float*)d_in[1];  // [D, E]
  const float* W1 = (const float*)d_in[2];  // [E, D, F]
  const float* b1 = (const float*)d_in[3];  // [E, F]
  const float* W2 = (const float*)d_in[4];  // [E, F, D]
  const float* b2 = (const float*)d_in[5];  // [E, D]
  float* out = (float*)d_out;

  char* ws = (char*)d_ws;
  size_t o = 0;
  auto take = [&](size_t bytes) -> void* {
    void* p = ws + o;
    o += (bytes + 255) & ~(size_t)255;
    return p;
  };
  int*   off        = (int*)take(NEXP * sizeof(int));
  int*   cnt        = (int*)take(NEXP * sizeof(int));
  int*   tok_e      = (int*)take((size_t)T_TOK * 2 * sizeof(int));
  float* tok_w      = (float*)take((size_t)T_TOK * 2 * sizeof(float));
  int*   tok2slot   = (int*)take((size_t)T_TOK * 2 * sizeof(int));
  int*   slot_token = (int*)take((size_t)T_TOK * 2 * sizeof(int));
  __hip_bfloat16* Xc  = (__hip_bfloat16*)take((size_t)2 * T_TOK * DDIM * sizeof(__hip_bfloat16)); // 8 MB
  __hip_bfloat16* Ac  = (__hip_bfloat16*)take((size_t)2 * T_TOK * FDIM * sizeof(__hip_bfloat16)); // 32 MB
  float*          Yc  = (float*)take((size_t)2 * T_TOK * DDIM * sizeof(float));                   // 16 MB
  __hip_bfloat16* W1t = (__hip_bfloat16*)take((size_t)NEXP * DDIM * FDIM * sizeof(__hip_bfloat16)); // 64 MB
  __hip_bfloat16* W2t = (__hip_bfloat16*)take((size_t)NEXP * DDIM * FDIM * sizeof(__hip_bfloat16)); // 64 MB

  // Weights fp32 -> bf16, transposed to N-major (once per call; ~16us of HBM).
  transpose_cvt_bf16<<<dim3(FDIM / 32, DDIM / 32, NEXP), dim3(32, 8), 0, stream>>>(W1, W1t, DDIM, FDIM);
  transpose_cvt_bf16<<<dim3(DDIM / 32, FDIM / 32, NEXP), dim3(32, 8), 0, stream>>>(W2, W2t, FDIM, DDIM);

  // Routing + deterministic compaction.
  moe_router<<<dim3(T_TOK), dim3(256), 0, stream>>>(x, Wr, tok_e, tok_w);
  moe_build<<<dim3(1), dim3(8), 0, stream>>>(tok_e, off, cnt, slot_token, tok2slot);
  moe_gather<<<dim3(2 * T_TOK), dim3(256), 0, stream>>>(x, slot_token, Xc);

  // Expert FFN: only routed rows (4096 total vs 16384 dense).
  moe_gemm_wmma<<<dim3(FDIM / 128, T_TOK / 128, NEXP), dim3(256), 0, stream>>>(
      Xc, W1t, b1, (void*)Ac, off, cnt, DDIM, FDIM, /*mode=*/0);
  moe_gemm_wmma<<<dim3(DDIM / 128, T_TOK / 128, NEXP), dim3(256), 0, stream>>>(
      Ac, W2t, b2, (void*)Yc, off, cnt, FDIM, DDIM, /*mode=*/1);

  moe_combine<<<dim3(T_TOK), dim3(256), 0, stream>>>(Yc, tok2slot, tok_w, out);

  (void)in_sizes; (void)n_in; (void)out_size; (void)ws_size;
}
